// GAT_35940286333219
// MI455X (gfx1250) — compile-verified, matrix operand order checked by reading, not run on hardware
//
#include <hip/hip_runtime.h>
#include <hip/hip_bf16.h>

// ---------------------------------------------------------------------------
// GATv2 stack for MI455X (gfx1250).
// Dense GEMMs: v_wmma_f32_16x16x32_bf16, 8-wave workgroups computing a
// 256x64 C macro-tile (32 rows x 64 cols per wave, 8 accumulators).
// B is staged through LDS (bf16, transposed [n][k]) with float4 global reads;
// each B fragment (2x ds_load_b128) feeds TWO back-to-back WMMAs (row tiles),
// and each A fragment is reused across 4 column tiles -> 8 WMMA / 32-K step.
// Edge softmax/aggregation: wave32-per-edge kernels with float atomics.
// ---------------------------------------------------------------------------

typedef __attribute__((ext_vector_type(16))) __bf16 v16bf;
typedef __attribute__((ext_vector_type(8)))  float  v8f;

#define NNODES  16384
#define NEDGES  262144
#define F_IN    128
#define DIM_    64
#define HEADS1  4
#define HID_    256
#define NEG_SLOPE 0.2f

#define BM 256   // C macro-tile rows per workgroup (8 waves x 32)
#define BN 64    // C macro-tile cols per workgroup (4 x 16 per wave)

// ---------------------------------------------------------------------------
// WMMA bf16 GEMM:  C[M,Nc] = act(A[M,K] @ B[K,Nc] + bias)
// Fragment layout per CDNA5 ISA 7.12.2:
//   A 16x32: lane l -> row l%16; k-halves by l/16 (khi 0/8); elements t=0..7
//            cover k = khi+t, t=8..15 cover k = 16+khi+(t-8).
//   B 32x16: symmetric (lane -> col l%16, same k striping).
//   C/D v8f: row = r + 8*(l>=16), col = l%16 within the 16x16 tile.
// Requirements (true at all call sites): M % 256 == 0, Nc % 64 == 0,
// K % 32 == 0.
// ---------------------------------------------------------------------------
__global__ __launch_bounds__(256) void k_gemm_bf16(
    const float* __restrict__ A, const float* __restrict__ B,
    const float* __restrict__ bias, float* __restrict__ C,
    int M, int K, int Nc, int relu) {
  __shared__ __bf16 lds_b[BN * 32];       // B tile, transposed [n][k], 4 KB

  const int tid  = threadIdx.x;
  const int lane = tid & 31;
  const int wave = tid >> 5;              // 0..7: 32-row slab within macro tile
  const int bm   = blockIdx.x;            // M / 256
  const int bn   = blockIdx.y;            // Nc / 64

  const int rbase = bm * BM + wave * 32;
  const int m0    = rbase + (lane & 15);        // row tile 0
  const int m1    = rbase + 16 + (lane & 15);   // row tile 1
  const int khi   = (lane >> 4) << 3;           // 0 or 8

  // B staging: each thread loads 8 contiguous floats of one k-row (2x float4)
  const int sk = tid >> 3;                // 0..31 : k row within tile
  const int sn = (tid & 7) * 8;           // 0..56 : n start (8 wide)

  v8f acc[2][4] = {};                     // [row tile][col tile]

  for (int k0 = 0; k0 < K; k0 += 32) {
    // ---- cooperatively stage B tile (32 x 64) into LDS, bf16 transposed ----
    __syncthreads();
    {
      const float* bp = B + (size_t)(k0 + sk) * Nc + bn * BN + sn;
      float4 q0 = *(const float4*)bp;             // global_load_b128
      float4 q1 = *(const float4*)(bp + 4);       // global_load_b128
      __bf16* lb = lds_b + sk;                    // [(sn+j)*32 + sk]
      lb[(sn + 0) * 32] = (__bf16)q0.x;
      lb[(sn + 1) * 32] = (__bf16)q0.y;
      lb[(sn + 2) * 32] = (__bf16)q0.z;
      lb[(sn + 3) * 32] = (__bf16)q0.w;
      lb[(sn + 4) * 32] = (__bf16)q1.x;
      lb[(sn + 5) * 32] = (__bf16)q1.y;
      lb[(sn + 6) * 32] = (__bf16)q1.z;
      lb[(sn + 7) * 32] = (__bf16)q1.w;
    }
    __syncthreads();

    // ---- two A fragments (fp32 -> bf16), contiguous 16B runs ----
    const float* ar0 = A + (size_t)m0 * K + k0 + khi;
    const float* ar1 = A + (size_t)m1 * K + k0 + khi;
    v16bf a0, a1;
#pragma unroll
    for (int t = 0; t < 8; ++t) {
      a0[t]     = (__bf16)ar0[t];
      a0[t + 8] = (__bf16)ar0[16 + t];
      a1[t]     = (__bf16)ar1[t];
      a1[t + 8] = (__bf16)ar1[16 + t];
    }
    if (k0 + 32 < K) {                    // global_prefetch_b8 for next A step
      __builtin_prefetch(ar0 + 32, 0, 1);
      __builtin_prefetch(ar1 + 32, 0, 1);
    }

    // ---- 8 WMMAs: each B fragment feeds both row tiles ----
#pragma unroll
    for (int c = 0; c < 4; ++c) {
      const __bf16* pb = &lds_b[(c * 16 + (lane & 15)) * 32 + khi];
      v16bf b;
#pragma unroll
      for (int t = 0; t < 8; ++t) {       // two 16B chunks -> ds_load_b128 x2
        b[t]     = pb[t];
        b[t + 8] = pb[16 + t];
      }
      acc[0][c] = __builtin_amdgcn_wmma_f32_16x16x32_bf16(
          false, a0, false, b, (short)0, acc[0][c], false, false);
      acc[1][c] = __builtin_amdgcn_wmma_f32_16x16x32_bf16(
          false, a1, false, b, (short)0, acc[1][c], false, false);
    }
  }

  // ---- epilogue: bias + optional relu ----
#pragma unroll
  for (int rt = 0; rt < 2; ++rt) {
    const int mbase = rbase + rt * 16 + ((lane >> 4) << 3);
#pragma unroll
    for (int c = 0; c < 4; ++c) {
      int ncol = bn * BN + c * 16 + (lane & 15);
      float bv = bias ? bias[ncol] : 0.0f;
#pragma unroll
      for (int r = 0; r < 8; ++r) {
        float v = acc[rt][c][r] + bv;
        if (relu) v = fmaxf(v, 0.0f);
        C[(size_t)(mbase + r) * Nc + ncol] = v;
      }
    }
  }
}

// ---------------------------------------------------------------------------
// Edge helpers
// ---------------------------------------------------------------------------
__device__ __forceinline__ void edge_src_dst(const int* __restrict__ ei,
                                             int e, int E, int& src, int& dst) {
  if (e < E) { src = ei[e]; dst = ei[E + e]; }
  else       { src = dst = e - E; }        // self loop
}

__device__ __forceinline__ void atomicMaxFloat(float* addr, float val) {
  // init pattern must be -inf (0xFF800000); works for mixed signs.
  if (val >= 0.0f) atomicMax((int*)addr, __float_as_int(val));
  else             atomicMin((unsigned int*)addr, __float_as_uint(val));
}

// m = -inf, denom = 0, agg[n,:] = bias  (HD is a power of two)
__global__ void k_layer_init(float* __restrict__ m, float* __restrict__ denom,
                             float* __restrict__ agg, const float* __restrict__ bias,
                             int n, int H, int HD) {
  int idx = blockIdx.x * blockDim.x + threadIdx.x;
  if (idx < n * H) { m[idx] = -__builtin_inff(); denom[idx] = 0.0f; }
  if (idx < n * HD) agg[idx] = bias[idx & (HD - 1)];
}

// logits[e,h] = sum_d leaky_relu(xl[src,h,d] + xr[dst,h,d]) * att[h,d]
// one wave per edge; segment-max into m[dst,h] via float atomic-max trick.
__global__ void k_edge_logits(const float* __restrict__ xl,
                              const float* __restrict__ xr,
                              const float* __restrict__ att,
                              const int* __restrict__ ei,
                              float* __restrict__ logits,
                              float* __restrict__ m,
                              int E, int n, int H, int dShift) {
  int e    = (blockIdx.x * blockDim.x + threadIdx.x) >> 5;
  int lane = threadIdx.x & 31;
  int etot = E + n;
  if (e >= etot) return;
  int src, dst;
  edge_src_dst(ei, e, E, src, dst);

  const int HD = H << dShift;
  float acch[HEADS1] = {0.0f, 0.0f, 0.0f, 0.0f};
  const float* pl = xl + (size_t)src * HD;
  const float* pr = xr + (size_t)dst * HD;
  for (int i = lane; i < HD; i += 32) {
    float v = pl[i] + pr[i];
    v = (v > 0.0f) ? v : NEG_SLOPE * v;
    acch[i >> dShift] += v * att[i];
  }
#pragma unroll
  for (int h = 0; h < HEADS1; ++h) {
    if (h >= H) break;
    float v = acch[h];
#pragma unroll
    for (int off = 16; off > 0; off >>= 1) v += __shfl_xor(v, off, 32);
    if (lane == 0) {
      logits[(size_t)e * H + h] = v;
      atomicMaxFloat(&m[dst * H + h], v);
    }
  }
}

// alpha = exp(logit - m[dst]); overwrite logits; denom[dst,h] += alpha
__global__ void k_edge_alpha(float* __restrict__ logits,
                             const float* __restrict__ m,
                             float* __restrict__ denom,
                             const int* __restrict__ ei,
                             int E, int n, int H) {
  int idx = blockIdx.x * blockDim.x + threadIdx.x;
  int tot = (E + n) * H;
  if (idx >= tot) return;
  int e = idx / H, h = idx - e * H;
  int dst = (e < E) ? ei[E + e] : e - E;
  float a = __expf(logits[idx] - m[dst * H + h]);
  logits[idx] = a;
  atomicAdd(&denom[dst * H + h], a);
}

// agg[dst,:] += (alpha/denom) * xl[src,:]   (one wave per edge)
__global__ void k_edge_aggregate(const float* __restrict__ alpha,
                                 const float* __restrict__ denom,
                                 const float* __restrict__ xl,
                                 const int* __restrict__ ei,
                                 float* __restrict__ agg,
                                 int E, int n, int H, int dShift) {
  int e    = (blockIdx.x * blockDim.x + threadIdx.x) >> 5;
  int lane = threadIdx.x & 31;
  int etot = E + n;
  if (e >= etot) return;
  int src, dst;
  edge_src_dst(ei, e, E, src, dst);

  const int HD = H << dShift;
  float w[HEADS1];
#pragma unroll
  for (int h = 0; h < HEADS1; ++h) {
    if (h >= H) break;
    w[h] = alpha[(size_t)e * H + h] / denom[dst * H + h];
  }
  const float* pl = xl + (size_t)src * HD;
  float* po = agg + (size_t)dst * HD;
  for (int i = lane; i < HD; i += 32)
    atomicAdd(&po[i], w[i >> dShift] * pl[i]);
}

// out = relu(agg) + add     (residual / skip connection)
__global__ void k_relu_add(const float* __restrict__ agg,
                           const float* __restrict__ add,
                           float* __restrict__ out, int n) {
  int i = blockIdx.x * blockDim.x + threadIdx.x;
  if (i < n) out[i] = fmaxf(agg[i], 0.0f) + add[i];
}

// out[n] = t2[n,:] . w[:,0] + b[0]
__global__ void k_final_matvec(const float* __restrict__ t2,
                               const float* __restrict__ w,
                               const float* __restrict__ b,
                               float* __restrict__ out, int n, int K) {
  int i = blockIdx.x * blockDim.x + threadIdx.x;
  if (i >= n) return;
  float s = b[0];
  const float* row = t2 + (size_t)i * K;
#pragma unroll 8
  for (int k = 0; k < K; ++k) s += row[k] * w[k];
  out[i] = s;
}

// ---------------------------------------------------------------------------
// Host orchestration
// ---------------------------------------------------------------------------
extern "C" void kernel_launch(void* const* d_in, const int* in_sizes, int n_in,
                              void* d_out, int out_size, void* d_ws, size_t ws_size,
                              hipStream_t stream) {
  (void)in_sizes; (void)n_in; (void)out_size; (void)ws_size;
  const float* x     = (const float*)d_in[0];
  const int*   ei    = (const int*)  d_in[1];
  const float* Win   = (const float*)d_in[2];
  const float* b_in  = (const float*)d_in[3];
  const float* Wskip = (const float*)d_in[4];
  const float* bskip = (const float*)d_in[5];
  const float* Wl1   = (const float*)d_in[6];
  const float* Wr1   = (const float*)d_in[7];
  const float* att1  = (const float*)d_in[8];
  const float* b1    = (const float*)d_in[9];
  const float* Wl2   = (const float*)d_in[10];
  const float* Wr2   = (const float*)d_in[11];
  const float* att2  = (const float*)d_in[12];
  const float* b2    = (const float*)d_in[13];
  const float* Wl3   = (const float*)d_in[14];
  const float* Wr3   = (const float*)d_in[15];
  const float* att3  = (const float*)d_in[16];
  const float* b3    = (const float*)d_in[17];
  const float* Wm1   = (const float*)d_in[18];
  const float* bm1   = (const float*)d_in[19];
  const float* Wm2   = (const float*)d_in[20];
  const float* bm2   = (const float*)d_in[21];
  const float* Wm3   = (const float*)d_in[22];
  const float* bm3   = (const float*)d_in[23];
  float* out = (float*)d_out;

  const int N = NNODES, E = NEDGES, ETOT = NEDGES + NNODES;

  // ---- workspace carve-up (fp32) ----
  float* ws = (float*)d_ws;
  float* h      = ws; ws += (size_t)N * DIM_;     // [N,64]
  float* skip   = ws; ws += (size_t)N * HID_;     // [N,256]
  float* xl     = ws; ws += (size_t)N * HID_;
  float* xr     = ws; ws += (size_t)N * HID_;
  float* agg    = ws; ws += (size_t)N * HID_;
  float* out1   = ws; ws += (size_t)N * HID_;
  float* out2   = ws; ws += (size_t)N * HID_;
  float* out3   = ws; ws += (size_t)N * HID_;
  float* logits = ws; ws += (size_t)ETOT * HEADS1;  // reused as alpha in-place
  float* mbuf   = ws; ws += (size_t)N * HEADS1;
  float* denom  = ws; ws += (size_t)N * HEADS1;
  float* t1 = xl;   // MLP temporaries reuse xl/xr after layer 3
  float* t2 = xr;

  auto cdiv = [](long a, long b) { return (int)((a + b - 1) / b); };

  auto gemm = [&](const float* A, const float* B, const float* bias, float* C,
                  int M, int K, int Nc, int relu) {
    dim3 grid(M / BM, Nc / BN);
    k_gemm_bf16<<<grid, 256, 0, stream>>>(A, B, bias, C, M, K, Nc, relu);
  };

  auto gat_layer = [&](const float* xin, int Kin,
                       const float* Wl, const float* Wr,
                       const float* att, const float* bias,
                       int H, int dShift,
                       const float* residual, float* outbuf) {
    gemm(xin, Wl, nullptr, xl, N, Kin, HID_, 0);
    gemm(xin, Wr, nullptr, xr, N, Kin, HID_, 0);
    k_layer_init<<<cdiv((long)N * HID_, 256), 256, 0, stream>>>(
        mbuf, denom, agg, bias, N, H, HID_);
    k_edge_logits<<<cdiv((long)ETOT * 32, 256), 256, 0, stream>>>(
        xl, xr, att, ei, logits, mbuf, E, N, H, dShift);
    k_edge_alpha<<<cdiv((long)ETOT * H, 256), 256, 0, stream>>>(
        logits, mbuf, denom, ei, E, N, H);
    k_edge_aggregate<<<cdiv((long)ETOT * 32, 256), 256, 0, stream>>>(
        logits, denom, xl, ei, agg, E, N, H, dShift);
    k_relu_add<<<cdiv((long)N * HID_, 256), 256, 0, stream>>>(
        agg, residual, outbuf, N * HID_);
  };

  // h = relu(x @ Win + b_in)                       [N,128] -> [N,64]
  gemm(x, Win, b_in, h, N, F_IN, DIM_, 1);
  // skip = h @ Wskip + bskip                       [N,64] -> [N,256]
  gemm(h, Wskip, bskip, skip, N, DIM_, HID_, 0);

  // layer 1: H=4, D=64 (dShift=6), out1 = relu(gat(h)) + skip
  gat_layer(h,    DIM_, Wl1, Wr1, att1, b1, HEADS1, 6, skip, out1);
  // layer 2: H=1, D=256 (dShift=8), out2 = relu(gat(out1)) + out1
  gat_layer(out1, HID_, Wl2, Wr2, att2, b2, 1, 8, out1, out2);
  // layer 3: out3 = relu(gat(out2)) + out2
  gat_layer(out2, HID_, Wl3, Wr3, att3, b3, 1, 8, out2, out3);

  // MLP head
  gemm(out3, Wm1, bm1, t1, N, HID_, DIM_, 1);      // [N,256] -> [N,64], relu
  gemm(t1,   Wm2, bm2, t2, N, DIM_, DIM_, 1);      // [N,64]  -> [N,64], relu
  k_final_matvec<<<cdiv((long)N, 256), 256, 0, stream>>>(t2, Wm3, bm3, out, N, DIM_);
}
